// MyAttention_89584427860075
// MI455X (gfx1250) — matrix operand order repaired and checked by reading, not compiled, not run on hardware
//
#include <hip/hip_runtime.h>

// MI455X (gfx1250) flash-attention:
//   q=in*qw, k=in*kw, v=in*vw (elementwise, broadcast over batch)
//   out = softmax(q k^T / sqrt(8)) v        B=8, L=2048, D=512
// Compute-bound (~69 GFLOP matmul vs ~80MB I/O) -> f16 WMMA path,
// with double-buffered GLOBAL_LOAD_ASYNC_TO_LDS tile prefetch (ASYNCcnt).

typedef __attribute__((ext_vector_type(16))) _Float16 v16h;
typedef __attribute__((ext_vector_type(8)))  _Float16 v8h;
typedef __attribute__((ext_vector_type(4)))  _Float16 v4h;
typedef __attribute__((ext_vector_type(8)))  float    v8f;
typedef __attribute__((ext_vector_type(4)))  float    v4f;

constexpr int kB = 8;
constexpr int kL = 2048;
constexpr int kD = 512;
constexpr int BM = 32;       // query rows per workgroup (2 row-groups of 16)
constexpr int BN = 32;       // key columns per iteration
constexpr int DSL = 128;     // D-slice per wave
constexpr int NTHREADS = 256;

typedef __attribute__((address_space(3))) const void* as3_cptr;
__device__ __forceinline__ unsigned lds_off(const void* p) {
  return (unsigned)(size_t)(as3_cptr)p;   // flat -> LDS address space
}

// ---------------- Pass 1: fused q/k/v projection, f32 -> f16 ----------------
__global__ __launch_bounds__(256) void qkv_project(
    const float* __restrict__ in, const float* __restrict__ qw,
    const float* __restrict__ kw, const float* __restrict__ vw,
    _Float16* __restrict__ qh, _Float16* __restrict__ kh,
    _Float16* __restrict__ vh)
{
  size_t g = (size_t)blockIdx.x * blockDim.x + threadIdx.x; // 1 thread : 4 elems
  size_t e = g * 4;
  size_t wofs = e % ((size_t)kL * kD);        // == l*D + d (weights are (L,D))
  v4f x = *(const v4f*)(in + e);
  v4f a = *(const v4f*)(qw + wofs);
  v4f b = *(const v4f*)(kw + wofs);
  v4f c = *(const v4f*)(vw + wofs);
  v4h hq, hk, hv;
#pragma unroll
  for (int t = 0; t < 4; ++t) {
    hq[t] = (_Float16)(x[t] * a[t]);
    hk[t] = (_Float16)(x[t] * b[t]);
    hv[t] = (_Float16)(x[t] * c[t]);
  }
  *(v4h*)(qh + e) = hq;
  *(v4h*)(kh + e) = hk;
  *(v4h*)(vh + e) = hv;
}

// ---- async tile fetch: global -> LDS, 16B per lane, no VGPR data staging ---
__device__ __forceinline__ void ktile_async(const _Float16* kb, int j0, int tid,
                                            _Float16 (*dst)[kD]) {
  unsigned long long base = (unsigned long long)(size_t)kb;  // block-uniform
#pragma unroll
  for (int r = 0; r < 8; ++r) {
    int idx = (r * NTHREADS + tid) * 8;     // halfs, < 32*512
    int kk = idx >> 9;
    int dd = idx & 511;
    unsigned goff = (unsigned)(((j0 + kk) * kD + dd) * (int)sizeof(_Float16));
    unsigned ldsa = lds_off(&dst[kk][dd]);
    asm volatile("global_load_async_to_lds_b128 %0, %1, %2"
                 :: "v"(ldsa), "v"(goff), "s"(base) : "memory");
  }
}

// ---- V tile: staged through VGPRs so it lands transposed (d-major) in LDS --
__device__ __forceinline__ void vtile_transpose(const _Float16* vb, int j0, int tid,
                                                _Float16 (*dst)[BN]) {
#pragma unroll
  for (int r = 0; r < 8; ++r) {
    int idx = (r * NTHREADS + tid) * 8;
    int kk = idx >> 9;
    int dd = idx & 511;
    v8h vv = *(const v8h*)(vb + (size_t)(j0 + kk) * kD + dd);
#pragma unroll
    for (int t = 0; t < 8; ++t) dst[dd + t][kk] = vv[t];
  }
}

// ---------------- Pass 2: flash attention with v_wmma_f32_16x16x32_f16 ------
__global__ __launch_bounds__(NTHREADS) void flash_attn(
    const _Float16* __restrict__ qh, const _Float16* __restrict__ kh,
    const _Float16* __restrict__ vh, float* __restrict__ out)
{
  __shared__ __align__(64) _Float16 Kt[2][BN][kD];  // 64 KB, row-major (key, d)
  __shared__ __align__(64) _Float16 Vt[2][kD][BN];  // 64 KB, transposed (d, key)
  __shared__ __align__(64) float    St[2][16][BN];  // 4 KB  S tiles per row-group
  __shared__ __align__(64) _Float16 Pt[2][16][BN];  // 2 KB  P (f16) per row-group
  __shared__ float At[2][16];                        // per-row rescale alpha
  __shared__ float Lt[2][16];                        // per-row softmax denom

  const int tid  = threadIdx.x;
  const int wave = tid >> 5;
  const int lane = tid & 31;
  const int rg   = wave >> 2;        // row-group (0/1): rows [rg*16, rg*16+16)
  const int dsl  = wave & 3;         // D-slice index
  const int d0   = dsl * DSL;
  const int hgrp = lane >> 4;        // half-lane group (wave32: two groups of 16)
  const int l16  = lane & 15;

  const int bidx = blockIdx.x / (kL / BM);
  const int mt   = blockIdx.x % (kL / BM);
  const int i0   = mt * BM + rg * 16;

  const _Float16* qb = qh + (size_t)bidx * kL * kD;
  const _Float16* kb = kh + (size_t)bidx * kL * kD;
  const _Float16* vb = vh + (size_t)bidx * kL * kD;

  // --- Q A-fragments for this wave's D-slice: 4 chunks of K=32 ---
  // A k-order used here (contiguous, B uses the same): lane(hgrp,m) element i -> K = i + 16*hgrp
  v16h qf[4];
  {
    const _Float16* qrow = qb + (size_t)(i0 + l16) * kD;
#pragma unroll
    for (int c = 0; c < 4; ++c) {
      int dbase = d0 + c * 32;
      v8h lo = *(const v8h*)(qrow + dbase + 16 * hgrp);
      v8h hi = *(const v8h*)(qrow + dbase + 8 + 16 * hgrp);
#pragma unroll
      for (int t = 0; t < 8; ++t) { qf[c][t] = lo[t]; qf[c][t + 8] = hi[t]; }
    }
  }

  v8f o[8];
#pragma unroll
  for (int c = 0; c < 8; ++c) o[c] = (v8f){};
  float mrun = -3.0e38f, lrun = 0.0f;
  const float scale = 0.35355339059327373f; // 1/sqrt(8)

  // --- prologue: bring tile 0 into buffer 0 ---
  ktile_async(kb, 0, tid, Kt[0]);
  vtile_transpose(vb, 0, tid, Vt[0]);
  asm volatile("s_wait_asynccnt 0" ::: "memory");
  __syncthreads();

  for (int j0 = 0, it = 0; j0 < kL; j0 += BN, ++it) {
    const int buf  = it & 1;
    const int nbuf = buf ^ 1;

    // [A] prefetch next tile into the other buffer (async K + staged V).
    // Safe: buffer nbuf was last read before the end-barrier of the previous
    // iteration; it is next read after barrier b0 of the next iteration.
    if (j0 + BN < kL) {
      ktile_async(kb, j0 + BN, tid, Kt[nbuf]);
      vtile_transpose(vb, j0 + BN, tid, Vt[nbuf]);
    }
    // zero S tiles (prior softmax reads finished before last barrier)
    {
      float* sp = &St[0][0][0];
      for (int t = tid; t < 2 * 16 * BN; t += NTHREADS) sp[t] = 0.0f;
    }
    __syncthreads();  // b0: St zeroed; current tiles resident

    // --- S partial = Q(slice) * K(slice)^T ; B layout: element i -> K=i+16*hgrp
    v8f c0 = (v8f){}, c1 = (v8f){};
#pragma unroll
    for (int c = 0; c < 4; ++c) {
      int db = d0 + c * 32 + 16 * hgrp;
      v16h bk0 = *(const v16h*)(&Kt[buf][l16][db]);        // keys 0..15
      v16h bk1 = *(const v16h*)(&Kt[buf][16 + l16][db]);   // keys 16..31
      c0 = __builtin_amdgcn_wmma_f32_16x16x32_f16(false, qf[c], false, bk0,
                                                  (short)0, c0, false, false);
      c1 = __builtin_amdgcn_wmma_f32_16x16x32_f16(false, qf[c], false, bk1,
                                                  (short)0, c1, false, false);
    }
    // reduce the 4 D-slice partials: C layout lane(hgrp,n) vgpr j -> M=j+8*hgrp
#pragma unroll
    for (int j = 0; j < 8; ++j) {
      atomicAdd(&St[rg][j + 8 * hgrp][l16],      c0[j]);
      atomicAdd(&St[rg][j + 8 * hgrp][16 + l16], c1[j]);
    }
    __syncthreads();  // b1: S complete

    // --- online softmax over this 32-key tile (redundant across D-slice waves)
    float vals[16];
    float mx = -3.0e38f;
#pragma unroll
    for (int i = 0; i < 16; ++i) {
      vals[i] = St[rg][l16][i + 16 * hgrp] * scale;
      mx = fmaxf(mx, vals[i]);
    }
    mx = fmaxf(mx, __shfl_xor(mx, 16, 32));   // combine the two half-row lanes
    float mnew = fmaxf(mrun, mx);
    float rs = 0.0f;
#pragma unroll
    for (int i = 0; i < 16; ++i) { vals[i] = __expf(vals[i] - mnew); rs += vals[i]; }
    rs += __shfl_xor(rs, 16, 32);
    float alpha = __expf(mrun - mnew);
    lrun = lrun * alpha + rs;
    mrun = mnew;
#pragma unroll
    for (int i = 0; i < 16; ++i) Pt[rg][l16][i + 16 * hgrp] = (_Float16)vals[i];
    if (hgrp == 0 && dsl == 0) At[rg][l16] = alpha;
    __syncthreads();  // b2: P / alpha published

    // --- rescale O and accumulate P * V(slice)
    float a8[8];
#pragma unroll
    for (int j = 0; j < 8; ++j) a8[j] = At[rg][j + 8 * hgrp];
#pragma unroll
    for (int c = 0; c < 8; ++c)
#pragma unroll
      for (int j = 0; j < 8; ++j) o[c][j] *= a8[j];

    v16h pf;
    {
      v8h lo = *(const v8h*)(&Pt[rg][l16][16 * hgrp]);
      v8h hi = *(const v8h*)(&Pt[rg][l16][8 + 16 * hgrp]);
#pragma unroll
      for (int t = 0; t < 8; ++t) { pf[t] = lo[t]; pf[t + 8] = hi[t]; }
    }
#pragma unroll
    for (int c = 0; c < 8; ++c) {
      v16h bv = *(const v16h*)(&Vt[buf][d0 + c * 16 + l16][16 * hgrp]);
      o[c] = __builtin_amdgcn_wmma_f32_16x16x32_f16(false, pf, false, bv,
                                                    (short)0, o[c], false, false);
    }

    // publish the async-prefetched tile before the next iteration reads it
    asm volatile("s_wait_asynccnt 0" ::: "memory");
    __syncthreads();  // b3: end of iteration
  }

  // --- finalize: divide by softmax denominator, write f32 output ---
  if (hgrp == 0 && dsl == 0) Lt[rg][l16] = lrun;
  __syncthreads();
  float inv8[8];
#pragma unroll
  for (int j = 0; j < 8; ++j) inv8[j] = 1.0f / Lt[rg][j + 8 * hgrp];
  float* ob = out + ((size_t)bidx * kL + i0) * kD;
#pragma unroll
  for (int c = 0; c < 8; ++c)
#pragma unroll
    for (int j = 0; j < 8; ++j)
      ob[(size_t)(j + 8 * hgrp) * kD + d0 + c * 16 + l16] = o[c][j] * inv8[j];
}

extern "C" void kernel_launch(void* const* d_in, const int* in_sizes, int n_in,
                              void* d_out, int out_size, void* d_ws, size_t ws_size,
                              hipStream_t stream) {
  const float* in = (const float*)d_in[0];
  const float* qw = (const float*)d_in[1];
  const float* kw = (const float*)d_in[2];
  const float* vw = (const float*)d_in[3];
  float* out = (float*)d_out;

  const size_t n = (size_t)kB * kL * kD; // 8M elements
  _Float16* qh = (_Float16*)d_ws;        // 16 MB
  _Float16* kh = qh + n;                 // 16 MB
  _Float16* vh = kh + n;                 // 16 MB  (48 MB total workspace)

  dim3 g1((unsigned)(n / 4 / 256));
  qkv_project<<<g1, 256, 0, stream>>>(in, qw, kw, vw, qh, kh, vh);

  dim3 g2((unsigned)(kB * (kL / BM)));   // 512 workgroups
  flash_attn<<<g2, NTHREADS, 0, stream>>>(qh, kh, vh, out);
}
